// Unet_52931176956384
// MI455X (gfx1250) — matrix operand order repaired and checked by reading, not compiled, not run on hardware
//
#include <hip/hip_runtime.h>

typedef __attribute__((ext_vector_type(16))) _Float16 v16h;
typedef __attribute__((ext_vector_type(2)))  _Float16 v2h;
typedef __attribute__((ext_vector_type(8)))  float    v8f;
typedef _Float16 h16;

#define BB  16
#define SS  32
#define NTT 2029
#define NN  256
#define EPSF 1e-5f

// ---------------------------------------------------------------------------
// 1) Beamforming: rec_unnorm[b,i,j] = mean_s sig[b,0,tbp[s,j,i],s]
//    (flip axis3 + rot90 == transpose, folded in by swapping i/j roles)
// ---------------------------------------------------------------------------
__global__ void k_beamform(const float* __restrict__ sig,
                           const float* __restrict__ trans,
                           float* __restrict__ bp) {
  int idx = blockIdx.x * 256 + threadIdx.x;
  if (idx >= BB * NN * NN) return;
  int b = idx >> 16;
  int r = idx & 65535;
  int i = r >> 8;
  int j = r & 255;
  const float Dx = 0.025f / (float)(NN - 1);
  const float x0 = -0.5f * (float)(NN - 1) * Dx;
  float xj = x0 + j * Dx;
  float yi = x0 + i * Dx;
  float acc = 0.f;
  for (int s = 0; s < SS; ++s) {
    float dx = xj - trans[2 * s];
    float dy = yi - trans[2 * s + 1];
    float d = sqrtf(dx * dx + dy * dy);
    float tf = d * (40000000.0f / 1572.0f) - 113.0f + 1.0f;
    float tr = fminf(fmaxf(roundf(tf), 0.f), (float)(NTT - 1));
    int ti = (int)tr;
    acc += sig[(b * NTT + ti) * SS + s];
  }
  bp[idx] = acc * (1.0f / (float)SS);
}

// ---------------------------------------------------------------------------
// 2) Global min/max + normalize
// ---------------------------------------------------------------------------
__global__ void k_minmax_part(const float* __restrict__ x, int n,
                              float* __restrict__ part) {
  __shared__ float smn[256], smx[256];
  int tid = threadIdx.x;
  float mn = 3.4e38f, mx = -3.4e38f;
  for (int i = blockIdx.x * 256 + tid; i < n; i += gridDim.x * 256) {
    float v = x[i];
    mn = fminf(mn, v);
    mx = fmaxf(mx, v);
  }
  smn[tid] = mn; smx[tid] = mx; __syncthreads();
  for (int s = 128; s > 0; s >>= 1) {
    if (tid < s) {
      smn[tid] = fminf(smn[tid], smn[tid + s]);
      smx[tid] = fmaxf(smx[tid], smx[tid + s]);
    }
    __syncthreads();
  }
  if (tid == 0) { part[2 * blockIdx.x] = smn[0]; part[2 * blockIdx.x + 1] = smx[0]; }
}

__global__ void k_minmax_final(const float* __restrict__ part, int nblk,
                               float* __restrict__ mm) {
  __shared__ float smn[256], smx[256];
  int tid = threadIdx.x;
  float mn = 3.4e38f, mx = -3.4e38f;
  for (int i = tid; i < nblk; i += 256) {
    mn = fminf(mn, part[2 * i]);
    mx = fmaxf(mx, part[2 * i + 1]);
  }
  smn[tid] = mn; smx[tid] = mx; __syncthreads();
  for (int s = 128; s > 0; s >>= 1) {
    if (tid < s) {
      smn[tid] = fminf(smn[tid], smn[tid + s]);
      smx[tid] = fmaxf(smx[tid], smx[tid + s]);
    }
    __syncthreads();
  }
  if (tid == 0) { mm[0] = smn[0]; mm[1] = smx[0]; }
}

__global__ void k_normalize(const float* __restrict__ bp,
                            const float* __restrict__ mm,
                            float* __restrict__ rec, int n) {
  int i = blockIdx.x * 256 + threadIdx.x;
  if (i >= n) return;
  rec[i] = (bp[i] - mm[0]) / (mm[1] - mm[0]);
}

// ---------------------------------------------------------------------------
// 3) First conv (Cin=1): scalar — K would be 94% padding under WMMA
// ---------------------------------------------------------------------------
__global__ void k_conv0(const float* __restrict__ rec,
                        const float* __restrict__ w,
                        float* __restrict__ out) {
  const int Ho = 128, Wo = 128, Co = 64;
  int idx = blockIdx.x * 256 + threadIdx.x;
  if (idx >= BB * Co * Ho * Wo) return;
  int ox = idx % Wo; int t = idx / Wo;
  int oy = t % Ho; t /= Ho;
  int co = t % Co; int b = t / Co;
  float acc = 0.f;
  for (int ky = 0; ky < 4; ++ky) {
    int iy = 2 * oy - 1 + ky;
    if (iy < 0 || iy >= 256) continue;
    for (int kx = 0; kx < 4; ++kx) {
      int ix = 2 * ox - 1 + kx;
      if (ix < 0 || ix >= 256) continue;
      acc += rec[(b * 256 + iy) * 256 + ix] * w[(co * 4 + ky) * 4 + kx];
    }
  }
  out[idx] = acc;
}

// ---------------------------------------------------------------------------
// 4) Staging kernels: zero-fill padded f16 buffers, f32 -> padded f16 copy,
//    plain f16 weight convert, deconv weight repack to [parity][n][K] f16
// ---------------------------------------------------------------------------
__global__ void k_zeroh(h16* __restrict__ p, int n) {
  int i = blockIdx.x * 256 + threadIdx.x;
  if (i < n) p[i] = (h16)0.f;
}

__global__ void k_cvtw(const float* __restrict__ x, h16* __restrict__ y, int n) {
  int i = blockIdx.x * 256 + threadIdx.x;
  if (i < n) y[i] = (h16)x[i];
}

__global__ void k_cvt_pad(const float* __restrict__ x, h16* __restrict__ y,
                          int C, int H) {
  int total = BB * C * H * H;
  int i = blockIdx.x * 256 + threadIdx.x;
  if (i >= total) return;
  int xx = i % H; int t = i / H;
  int yy = t % H; t /= H;
  int c = t % C; int b = t / C;
  int Hp = H + 2;
  y[((size_t)(b * C + c) * Hp + yy + 1) * Hp + xx + 1] = (h16)x[i];
}

__global__ void k_wrep(const float* __restrict__ w, h16* __restrict__ wr,
                       int Cin, int Cout) {
  int K = Cin * 4;
  int total = 4 * Cout * K;
  int idx = blockIdx.x * 256 + threadIdx.x;
  if (idx >= total) return;
  int k = idx % K; int t2 = idx / K;
  int n = t2 % Cout; int par = t2 / Cout;
  int py = par >> 1, px = par & 1;
  int ci = k >> 2, t = k & 3;
  int ky = py + 2 * (t >> 1), kx = px + 2 * (t & 1);
  wr[idx] = (h16)w[((ci * Cout + n) * 4 + (3 - ky)) * 4 + (3 - kx)];
}

// ---------------------------------------------------------------------------
// 5) WMMA implicit-GEMM stride-2 conv, 2x2 register tiling (32 px x 32 ch).
//    Compile-time shapes; halo-padded f16 inputs -> check-free vector gathers.
// ---------------------------------------------------------------------------
template<int Cin, int Hi, int Cout>
__global__ void k_conv_wmma(const h16* __restrict__ in,
                            const h16* __restrict__ wf,
                            float* __restrict__ out) {
  constexpr int Ho = Hi / 2, Wo = Hi / 2;
  constexpr int Hp = Hi + 2, Wp = Hi + 2, HWp = Hp * Wp;
  constexpr int K = Cin * 16;
  constexpr int tiles_m = (BB * Ho * Wo) / 32;
  constexpr int tiles_n = Cout / 32;
  int lane = threadIdx.x & 31;
  int gw = blockIdx.x * 8 + (threadIdx.x >> 5);
  if (gw >= tiles_m * tiles_n) return;          // wave-uniform
  int tm = gw % tiles_m, tn = gw / tiles_m;
  int half = lane >> 4;
  int lr = lane & 15;
  const h16* abase[2];
#pragma unroll
  for (int j = 0; j < 2; ++j) {
    int m = tm * 32 + j * 16 + lr;
    int bb = m / (Ho * Wo); int rm = m % (Ho * Wo);   // shifts (consts)
    int oy = rm / Wo, ox = rm % Wo;
    // padded window: rows 2*oy + (cl*2 + ry), cols 2*ox + kx; channel ci0 + half
    abase[j] = in + ((size_t)bb * Cin * Hp + 2 * oy) * Wp + 2 * ox;
  }
  int n0 = tn * 32 + lr;
  const h16* wp0 = wf + (size_t)n0 * K + half * 16;        // 32B aligned
  const h16* wp1 = wf + (size_t)(n0 + 16) * K + half * 16;
  v8f acc00 = {}, acc01 = {}, acc10 = {}, acc11 = {};
#pragma unroll 1
  for (int k0 = 0; k0 < K; k0 += 32) {
    if (k0 + 32 < K) {                         // cover latency on weights
      __builtin_prefetch(wp0 + k0 + 32, 0, 3);
      __builtin_prefetch(wp1 + k0 + 32, 0, 3);
    }
    int ci0 = k0 >> 4;                         // chunk covers ci0, ci0+1
    v16h a[2];
#pragma unroll
    for (int j = 0; j < 2; ++j) {
      const h16* p = abase[j] + (size_t)(ci0 + half) * HWp;
#pragma unroll
      for (int cl = 0; cl < 2; ++cl) {
#pragma unroll
        for (int ry = 0; ry < 2; ++ry) {
          const h16* q = p + (cl * 2 + ry) * Wp;
          v2h x0 = *(const v2h*)(q);           // 4B aligned (even offsets)
          v2h x1 = *(const v2h*)(q + 2);
          h16 vv[4] = {x0.x, x0.y, x1.x, x1.y};
#pragma unroll
          for (int kx = 0; kx < 4; ++kx) {
            h16 v = vv[kx];
            v = (v >= (h16)0.f) ? v : (h16)(v * (h16)0.2f);  // lrelu f16
            a[j][cl * 8 + ry * 4 + kx] = v;
          }
        }
      }
    }
    v16h bf0 = *(const v16h*)(wp0 + k0);
    v16h bf1 = *(const v16h*)(wp1 + k0);
    acc00 = __builtin_amdgcn_wmma_f32_16x16x32_f16(false, a[0], false, bf0, (short)0, acc00, false, false);
    acc01 = __builtin_amdgcn_wmma_f32_16x16x32_f16(false, a[0], false, bf1, (short)0, acc01, false, false);
    acc10 = __builtin_amdgcn_wmma_f32_16x16x32_f16(false, a[1], false, bf0, (short)0, acc10, false, false);
    acc11 = __builtin_amdgcn_wmma_f32_16x16x32_f16(false, a[1], false, bf1, (short)0, acc11, false, false);
  }
  auto store_tile = [&](const v8f& A, int j, int i) {
#pragma unroll
    for (int r = 0; r < 8; ++r) {
      int ms = tm * 32 + j * 16 + r + half * 8;
      int bs = ms / (Ho * Wo); int rs = ms % (Ho * Wo);
      int nn = tn * 32 + i * 16 + lr;
      out[((size_t)(bs * Cout + nn) * Ho + rs / Wo) * Wo + rs % Wo] = A[r];
    }
  };
  store_tile(acc00, 0, 0); store_tile(acc01, 0, 1);
  store_tile(acc10, 1, 0); store_tile(acc11, 1, 1);
}

// ---------------------------------------------------------------------------
// 6) WMMA transposed conv, parity-decomposed (blockIdx.y), 2x2 tiling.
//    K = Cin*4; per-chunk concat-side hoist (C1 % 8 == 0); padded inputs.
// ---------------------------------------------------------------------------
template<int C1, int C2, int Hi, int Cout>
__global__ void k_deconv_wmma(const h16* __restrict__ in1,
                              const h16* __restrict__ in2,
                              const h16* __restrict__ wr,
                              float* __restrict__ out) {
  constexpr int Cin = C1 + C2, K = Cin * 4;
  constexpr int Wi = Hi, Ho = Hi * 2, Wo = Hi * 2;
  constexpr int Hp = Hi + 2, Wp = Hi + 2, HWp = Hp * Wp;
  constexpr int tiles_m = (BB * Hi * Wi) / 32;
  constexpr int tiles_n = Cout / 32;
  int par = blockIdx.y; int py = par >> 1, px = par & 1;
  int lane = threadIdx.x & 31;
  int gw = blockIdx.x * 8 + (threadIdx.x >> 5);
  if (gw >= tiles_m * tiles_n) return;          // wave-uniform
  int tm = gw % tiles_m, tn = gw / tiles_m;
  int half = lane >> 4;
  int lr = lane & 15;
  const h16 *a1base[2], *a2base[2];
#pragma unroll
  for (int j = 0; j < 2; ++j) {
    int m = tm * 32 + j * 16 + lr;
    int bb = m / (Hi * Wi); int rm = m % (Hi * Wi);
    int y2 = rm / Wi, x2 = rm % Wi;
    size_t sp = ((size_t)(y2 + py)) * Wp + (x2 + px);   // padded 2x2 top-left
    a1base[j] = in1 + (size_t)bb * C1 * HWp + sp;
    a2base[j] = in2 + (size_t)bb * C2 * HWp + sp;
  }
  int n0 = tn * 32 + lr;
  const h16* wp0 = wr + ((size_t)par * Cout + n0) * K + half * 16;
  const h16* wp1 = wr + ((size_t)par * Cout + n0 + 16) * K + half * 16;
  v8f acc00 = {}, acc01 = {}, acc10 = {}, acc11 = {};
#pragma unroll 1
  for (int k0 = 0; k0 < K; k0 += 32) {
    if (k0 + 32 < K) {
      __builtin_prefetch(wp0 + k0 + 32, 0, 3);
      __builtin_prefetch(wp1 + k0 + 32, 0, 3);
    }
    int cb = k0 >> 2;                 // chunk spans ci in [cb, cb+7], one side
    v16h a[2];
#pragma unroll
    for (int j = 0; j < 2; ++j) {
      const h16* p = (cb < C1) ? a1base[j] + (size_t)cb * HWp
                               : a2base[j] + (size_t)(cb - C1) * HWp;
      p += (size_t)(half * 4) * HWp;
#pragma unroll
      for (int cl = 0; cl < 2; ++cl) {
#pragma unroll
        for (int cg = 0; cg < 2; ++cg) {        // ci offset group: +0 / +2
          const h16* q = p + (size_t)(cl + cg * 2) * HWp;
#pragma unroll
          for (int ty = 0; ty < 2; ++ty) {
#pragma unroll
            for (int tx = 0; tx < 2; ++tx) {
              h16 v = q[ty * Wp + tx];
              v = (v > (h16)0.f) ? v : (h16)0.f;          // relu f16
              a[j][(cg * 2 + cl) * 4 + ty * 2 + tx] = v;
            }
          }
        }
      }
    }
    v16h bf0 = *(const v16h*)(wp0 + k0);
    v16h bf1 = *(const v16h*)(wp1 + k0);
    acc00 = __builtin_amdgcn_wmma_f32_16x16x32_f16(false, a[0], false, bf0, (short)0, acc00, false, false);
    acc01 = __builtin_amdgcn_wmma_f32_16x16x32_f16(false, a[0], false, bf1, (short)0, acc01, false, false);
    acc10 = __builtin_amdgcn_wmma_f32_16x16x32_f16(false, a[1], false, bf0, (short)0, acc10, false, false);
    acc11 = __builtin_amdgcn_wmma_f32_16x16x32_f16(false, a[1], false, bf1, (short)0, acc11, false, false);
  }
  auto store_tile = [&](const v8f& A, int j, int i) {
#pragma unroll
    for (int r = 0; r < 8; ++r) {
      int ms = tm * 32 + j * 16 + r + half * 8;
      int bs = ms / (Hi * Wi); int rs = ms % (Hi * Wi);
      int oys = 2 * (rs / Wi) + py, oxs = 2 * (rs % Wi) + px;
      int nn = tn * 32 + i * 16 + lr;
      out[((size_t)(bs * Cout + nn) * Ho + oys) * Wo + oxs] = A[r];
    }
  };
  store_tile(acc00, 0, 0); store_tile(acc01, 0, 1);
  store_tile(acc10, 1, 0); store_tile(acc11, 1, 1);
}

// ---------------------------------------------------------------------------
// 7) Batch-norm: per-channel stats; apply writes f32 in-place (+ padded f16)
// ---------------------------------------------------------------------------
__global__ void k_bnstats(const float* __restrict__ x, float* __restrict__ st,
                          int C, int HW) {
  __shared__ float s1[256], s2[256];
  int c = blockIdx.x; int tid = threadIdx.x;
  int cnt = BB * HW;
  float a = 0.f, q = 0.f;
  for (int i = tid; i < cnt; i += 256) {
    int bb = i / HW; int p = i % HW;
    float v = x[(bb * C + c) * HW + p];
    a += v; q += v * v;
  }
  s1[tid] = a; s2[tid] = q; __syncthreads();
  for (int s = 128; s > 0; s >>= 1) {
    if (tid < s) { s1[tid] += s1[tid + s]; s2[tid] += s2[tid + s]; }
    __syncthreads();
  }
  if (tid == 0) {
    float mean = s1[0] / (float)cnt;
    float var = s2[0] / (float)cnt - mean * mean;
    st[2 * c] = mean;
    st[2 * c + 1] = rsqrtf(var + EPSF);
  }
}

__global__ void k_bnapply_pad(float* __restrict__ x, h16* __restrict__ xh,
                              const float* __restrict__ st,
                              const float* __restrict__ g,
                              const float* __restrict__ be,
                              int C, int H) {
  int total = BB * C * H * H;
  int i = blockIdx.x * 256 + threadIdx.x;
  if (i >= total) return;
  int xx = i % H; int t = i / H;
  int yy = t % H; t /= H;
  int c = t % C; int b = t / C;
  float v = (x[i] - st[2 * c]) * st[2 * c + 1] * g[c] + be[c];
  x[i] = v;
  int Hp = H + 2;
  xh[((size_t)(b * C + c) * Hp + yy + 1) * Hp + xx + 1] = (h16)v;
}

__global__ void k_bnapply(float* __restrict__ x, const float* __restrict__ st,
                          const float* __restrict__ g, const float* __restrict__ be,
                          int C, int HW, int total) {
  int i = blockIdx.x * 256 + threadIdx.x;
  if (i >= total) return;
  int c = (i / HW) % C;
  x[i] = (x[i] - st[2 * c]) * st[2 * c + 1] * g[c] + be[c];
}

// ---------------------------------------------------------------------------
// 8) Last deconv (Cout=1): scalar, relu-concat + bias + sigmoid fused
// ---------------------------------------------------------------------------
__global__ void k_final(const float* __restrict__ h0, const float* __restrict__ u1,
                        const float* __restrict__ w, const float* __restrict__ b0,
                        float* __restrict__ out) {
  int idx = blockIdx.x * 256 + threadIdx.x;
  if (idx >= BB * 256 * 256) return;
  int ox = idx & 255; int t = idx >> 8;
  int oy = t & 255; int b = t >> 8;
  int py = oy & 1, px = ox & 1;
  float acc = 0.f;
  for (int ci = 0; ci < 128; ++ci) {
    const float* src = (ci < 64) ? h0 + ((size_t)(b * 64 + ci)) * 128 * 128
                                 : u1 + ((size_t)(b * 64 + (ci - 64))) * 128 * 128;
    for (int ty = 0; ty < 2; ++ty) {
      int ky = py + 2 * ty; int ny = oy + ky - 2;
      if (ny < 0) continue; int iy = ny >> 1; if (iy >= 128) continue;
      for (int tx = 0; tx < 2; ++tx) {
        int kx = px + 2 * tx; int nx = ox + kx - 2;
        if (nx < 0) continue; int ix = nx >> 1; if (ix >= 128) continue;
        float v = src[iy * 128 + ix];
        v = (v > 0.f) ? v : 0.f;
        acc += v * w[ci * 16 + (3 - ky) * 4 + (3 - kx)];
      }
    }
  }
  acc += b0[0];
  out[idx] = 1.f / (1.f + __expf(-acc));
}

// ---------------------------------------------------------------------------
static inline int cdiv(int a, int b) { return (a + b - 1) / b; }

extern "C" void kernel_launch(void* const* d_in, const int* in_sizes, int n_in,
                              void* d_out, int out_size, void* d_ws, size_t ws_size,
                              hipStream_t stream) {
  const float* sig   = (const float*)d_in[0];
  const float* trans = (const float*)d_in[1];
  const float* w_d0  = (const float*)d_in[2];
  const float* w_d1  = (const float*)d_in[3];
  const float* g_d1  = (const float*)d_in[4];
  const float* b_d1  = (const float*)d_in[5];
  const float* w_d2  = (const float*)d_in[6];
  const float* g_d2  = (const float*)d_in[7];
  const float* b_d2  = (const float*)d_in[8];
  const float* w_d3  = (const float*)d_in[9];
  const float* g_d3  = (const float*)d_in[10];
  const float* b_d3  = (const float*)d_in[11];
  const float* w_d4  = (const float*)d_in[12];
  const float* w_u4  = (const float*)d_in[13];
  const float* g_u4  = (const float*)d_in[14];
  const float* b_u4  = (const float*)d_in[15];
  const float* w_u3  = (const float*)d_in[16];
  const float* g_u3  = (const float*)d_in[17];
  const float* b_u3  = (const float*)d_in[18];
  const float* w_u2  = (const float*)d_in[19];
  const float* g_u2  = (const float*)d_in[20];
  const float* b_u2  = (const float*)d_in[21];
  const float* w_u1  = (const float*)d_in[22];
  const float* g_u1  = (const float*)d_in[23];
  const float* b_u1  = (const float*)d_in[24];
  const float* w_u0  = (const float*)d_in[25];
  const float* b_u0  = (const float*)d_in[26];

  char* base = (char*)d_ws;
  size_t off = 0;
  auto allocf = [&](size_t n) { float* p = (float*)(base + off); off = (off + n * 4 + 63) & ~(size_t)63; return p; };
  auto alloch = [&](size_t n) { h16*  p = (h16*) (base + off); off = (off + n * 2 + 63) & ~(size_t)63; return p; };

  float* bp    = allocf(1048576);
  float* part  = allocf(8192);
  float* mm    = allocf(2);
  float* stats = allocf(1024);
  float* h0 = allocf(16777216);   // 16x64x128x128
  float* h1 = allocf(8388608);    // 16x128x64x64
  float* h2 = allocf(4194304);    // 16x256x32x32
  float* h3 = allocf(2097152);    // 16x512x16x16
  float* h4 = allocf(524288);     // 16x512x8x8
  float* u4 = allocf(2097152);
  float* u3 = allocf(4194304);
  float* u2 = allocf(8388608);
  float* u1 = allocf(16777216);

  // halo-padded f16 mirrors: (B, C, H+2, W+2)
  const int n_h0f = 16 * 64 * 130 * 130;   // 17,305,600
  const int n_h1f = 16 * 128 * 66 * 66;    //  8,921,088
  const int n_h2f = 16 * 256 * 34 * 34;    //  4,734,976
  const int n_h3f = 16 * 512 * 18 * 18;    //  2,654,208
  const int n_h4f = 16 * 512 * 10 * 10;    //    819,200
  h16* h0f = alloch(n_h0f);
  h16* h1f = alloch(n_h1f);
  h16* h2f = alloch(n_h2f);
  h16* h3f = alloch(n_h3f);
  h16* h4f = alloch(n_h4f);
  h16* u4f = alloch(n_h3f);
  h16* u3f = alloch(n_h2f);
  h16* u2f = alloch(n_h1f);

  h16* wd1f = alloch(131072);     // 128 x 1024
  h16* wd2f = alloch(524288);     // 256 x 2048
  h16* wd3f = alloch(2097152);    // 512 x 4096
  h16* wd4f = alloch(4194304);    // 512 x 8192
  h16* wu4r = alloch(4194304);    // 4 x 512 x 2048
  h16* wu3r = alloch(4194304);    // 4 x 256 x 4096
  h16* wu2r = alloch(1048576);    // 4 x 128 x 2048
  h16* wu1r = alloch(262144);     // 4 x 64 x 1024

  float* outp = (float*)d_out;
  float* rec  = outp + 1048576;   // tuple: (out, rec)

  // Zero padded buffers (borders stay 0 = lrelu(0) = relu(0))
  k_zeroh<<<cdiv(n_h0f, 256), 256, 0, stream>>>(h0f, n_h0f);
  k_zeroh<<<cdiv(n_h1f, 256), 256, 0, stream>>>(h1f, n_h1f);
  k_zeroh<<<cdiv(n_h2f, 256), 256, 0, stream>>>(h2f, n_h2f);
  k_zeroh<<<cdiv(n_h3f, 256), 256, 0, stream>>>(h3f, n_h3f);
  k_zeroh<<<cdiv(n_h4f, 256), 256, 0, stream>>>(h4f, n_h4f);
  k_zeroh<<<cdiv(n_h3f, 256), 256, 0, stream>>>(u4f, n_h3f);
  k_zeroh<<<cdiv(n_h2f, 256), 256, 0, stream>>>(u3f, n_h2f);
  k_zeroh<<<cdiv(n_h1f, 256), 256, 0, stream>>>(u2f, n_h1f);

  // Weight precision staging (conv weights are already GEMM-contiguous)
  k_cvtw<<<cdiv(131072, 256), 256, 0, stream>>>(w_d1, wd1f, 131072);
  k_cvtw<<<cdiv(524288, 256), 256, 0, stream>>>(w_d2, wd2f, 524288);
  k_cvtw<<<cdiv(2097152, 256), 256, 0, stream>>>(w_d3, wd3f, 2097152);
  k_cvtw<<<cdiv(4194304, 256), 256, 0, stream>>>(w_d4, wd4f, 4194304);
  k_wrep<<<cdiv(4194304, 256), 256, 0, stream>>>(w_u4, wu4r, 512, 512);
  k_wrep<<<cdiv(4194304, 256), 256, 0, stream>>>(w_u3, wu3r, 1024, 256);
  k_wrep<<<cdiv(1048576, 256), 256, 0, stream>>>(w_u2, wu2r, 512, 128);
  k_wrep<<<cdiv(262144, 256), 256, 0, stream>>>(w_u1, wu1r, 256, 64);

  // Beamforming + normalization
  k_beamform<<<cdiv(1048576, 256), 256, 0, stream>>>(sig, trans, bp);
  k_minmax_part<<<4096, 256, 0, stream>>>(bp, 1048576, part);
  k_minmax_final<<<1, 256, 0, stream>>>(part, 4096, mm);
  k_normalize<<<cdiv(1048576, 256), 256, 0, stream>>>(bp, mm, rec, 1048576);

  // Encoder
  k_conv0<<<cdiv(16 * 64 * 128 * 128, 256), 256, 0, stream>>>(rec, w_d0, h0);
  k_cvt_pad<<<cdiv(16777216, 256), 256, 0, stream>>>(h0, h0f, 64, 128);

  auto bnp = [&](float* x, h16* xh, const float* g, const float* be, int C, int H) {
    k_bnstats<<<C, 256, 0, stream>>>(x, stats, C, H * H);
    k_bnapply_pad<<<cdiv(BB * C * H * H, 256), 256, 0, stream>>>(x, xh, stats, g, be, C, H);
  };

  k_conv_wmma<64, 128, 128><<<cdiv(2048 * 4, 8), 256, 0, stream>>>(h0f, wd1f, h1);
  bnp(h1, h1f, g_d1, b_d1, 128, 64);
  k_conv_wmma<128, 64, 256><<<cdiv(512 * 8, 8), 256, 0, stream>>>(h1f, wd2f, h2);
  bnp(h2, h2f, g_d2, b_d2, 256, 32);
  k_conv_wmma<256, 32, 512><<<cdiv(128 * 16, 8), 256, 0, stream>>>(h2f, wd3f, h3);
  bnp(h3, h3f, g_d3, b_d3, 512, 16);
  k_conv_wmma<512, 16, 512><<<cdiv(32 * 16, 8), 256, 0, stream>>>(h3f, wd4f, h4);  // no BN
  k_cvt_pad<<<cdiv(524288, 256), 256, 0, stream>>>(h4, h4f, 512, 8);

  // Decoder (relu + concat fused in deconv loads)
  k_deconv_wmma<512, 0, 8, 512><<<dim3(cdiv(32 * 16, 8), 4), 256, 0, stream>>>(h4f, h4f, wu4r, u4);
  bnp(u4, u4f, g_u4, b_u4, 512, 16);
  k_deconv_wmma<512, 512, 16, 256><<<dim3(cdiv(128 * 8, 8), 4), 256, 0, stream>>>(h3f, u4f, wu3r, u3);
  bnp(u3, u3f, g_u3, b_u3, 256, 32);
  k_deconv_wmma<256, 256, 32, 128><<<dim3(cdiv(512 * 4, 8), 4), 256, 0, stream>>>(h2f, u3f, wu2r, u2);
  bnp(u2, u2f, g_u2, b_u2, 128, 64);
  k_deconv_wmma<128, 128, 64, 64><<<dim3(cdiv(2048 * 2, 8), 4), 256, 0, stream>>>(h1f, u2f, wu1r, u1);
  k_bnstats<<<64, 256, 0, stream>>>(u1, stats, 64, 16384);
  k_bnapply<<<cdiv(16777216, 256), 256, 0, stream>>>(u1, stats, g_u1, b_u1, 64, 16384, 16777216);

  k_final<<<cdiv(1048576, 256), 256, 0, stream>>>(h0, u1, w_u0, b_u0, outp);
}
